// ChannelAttentionModule_83829171683588
// MI455X (gfx1250) — matrix operand order repaired and checked by reading, not compiled, not run on hardware
//
#include <hip/hip_runtime.h>

#define BATCH   8
#define CCH     512
#define NPIX    16384
#define NSLICE  8
#define SLICE_N (NPIX / NSLICE)   // 2048
#define LDSTRIDE 40               // bf16 elems per LDS row (16B-aligned rows, bank-stagger)

typedef __attribute__((ext_vector_type(16))) __bf16        v16bf;
typedef __attribute__((ext_vector_type(8)))  float         v8f;
typedef __attribute__((ext_vector_type(4)))  unsigned int  u32x4;

// Pack two f32 -> two bf16 (RNE) in ONE VALU op. Inline asm pins the operand
// pairing so the backend cannot re-associate into cvt+byte-shuffle sequences.
__device__ __forceinline__ unsigned int pack2bf(float lo, float hi) {
    unsigned int r;
    asm("v_cvt_pk_bf16_f32 %0, %1, %2" : "=v"(r) : "v"(lo), "v"(hi));
    return r;
}

__device__ __forceinline__ unsigned short f2bf(float f) {
    return (unsigned short)(pack2bf(f, 0.f) & 0xFFFFu);
}

// Build a 16xbf16 fragment from two 16-byte chunks (LDS or global).
__device__ __forceinline__ v16bf make_frag(const unsigned short* p0, const unsigned short* p1) {
    union { v16bf v; u32x4 q[2]; } u;
    u.q[0] = *(const u32x4*)p0;
    u.q[1] = *(const u32x4*)p1;
    return u.v;
}

__device__ __forceinline__ v8f v8f_zero() {
    v8f z = {0.f, 0.f, 0.f, 0.f, 0.f, 0.f, 0.f, 0.f};
    return z;
}

// ---------------------------------------------------------------------------
// Kernel 1: partial gram, upper-triangle block tiles only (matrix is symmetric)
//   P[slice][b][c][d] = sum_{n in slice} q[b,n,c]*q[b,n,d]
// grid: (10, 1, BATCH*NSLICE), block 256 (8 waves). 128x128 tile per WG.
// Each WG writes its tile transposed (vectorized b128) at block (J,I); off-diag
// tiles additionally write the natural orientation at block (I,J).
// ---------------------------------------------------------------------------
__global__ __launch_bounds__(256)
void gram_kernel(const float* __restrict__ q, float* __restrict__ partials) {
    __shared__ __align__(16) unsigned short ldsA[128 * LDSTRIDE]; // [c - ci][n]
    __shared__ __align__(16) unsigned short ldsB[128 * LDSTRIDE]; // [c - cj][n]

    // decode upper-triangle pair index 0..9 -> (I <= J), I,J in 0..3
    int rem = blockIdx.x, I = 0;
    while (rem >= 4 - I) { rem -= 4 - I; ++I; }
    const int J = I + rem;
    const int ci = I * 128;
    const int cj = J * 128;
    const bool diag = (I == J);

    const int b     = blockIdx.z >> 3;
    const int slice = blockIdx.z & 7;

    const float* qb = q + (size_t)b * NPIX * CCH;
    float* part = partials + ((size_t)(slice * BATCH + b)) * CCH * CCH;

    const int tid  = threadIdx.x;
    const int lane = tid & 31;
    const int wid  = tid >> 5;
    const int wr   = wid & 3;          // 4 row groups of 32
    const int wc   = wid >> 2;         // 2 col groups of 64
    const int sel  = (lane >> 4) & 1;
    const int l15  = lane & 15;

    v8f acc[2][4];
    for (int r = 0; r < 2; ++r)
        for (int t = 0; t < 4; ++t)
            acc[r][t] = v8f_zero();

    const int nbase0 = slice * SLICE_N;
    for (int kc = 0; kc < SLICE_N; kc += 32) {
        const int nb = nbase0 + kc;
        __syncthreads();
        // Stage q[nb:nb+32, ci:+128] (and cj block if off-diag) transposed to
        // [c][n] bf16. Two adjacent n rows per thread -> packed b32 LDS stores;
        // each global float4 load is wave-contiguous (512B).
        for (int i = 0; i < 2; ++i) {
            int idx = tid + (i << 8);         // 0..511
            int n   = (idx >> 5) << 1;        // 0,2,..,30
            int c0  = (idx & 31) << 2;        // 0..124
            const float* s0 = qb + (size_t)(nb + n) * CCH + ci + c0;
            float4 a0 = *(const float4*)s0;
            float4 a1 = *(const float4*)(s0 + CCH);
            *(unsigned int*)&ldsA[(c0 + 0) * LDSTRIDE + n] = pack2bf(a0.x, a1.x);
            *(unsigned int*)&ldsA[(c0 + 1) * LDSTRIDE + n] = pack2bf(a0.y, a1.y);
            *(unsigned int*)&ldsA[(c0 + 2) * LDSTRIDE + n] = pack2bf(a0.z, a1.z);
            *(unsigned int*)&ldsA[(c0 + 3) * LDSTRIDE + n] = pack2bf(a0.w, a1.w);
            if (!diag) {
                const float* s1 = qb + (size_t)(nb + n) * CCH + cj + c0;
                float4 b0 = *(const float4*)s1;
                float4 b1 = *(const float4*)(s1 + CCH);
                *(unsigned int*)&ldsB[(c0 + 0) * LDSTRIDE + n] = pack2bf(b0.x, b1.x);
                *(unsigned int*)&ldsB[(c0 + 1) * LDSTRIDE + n] = pack2bf(b0.y, b1.y);
                *(unsigned int*)&ldsB[(c0 + 2) * LDSTRIDE + n] = pack2bf(b0.z, b1.z);
                *(unsigned int*)&ldsB[(c0 + 3) * LDSTRIDE + n] = pack2bf(b0.w, b1.w);
            }
        }
        __syncthreads();

        const unsigned short* Bt = diag ? ldsA : ldsB;
        v16bf afrag[2], bfrag[4];
        for (int r = 0; r < 2; ++r) {
            const unsigned short* base = ldsA + (wr * 32 + r * 16 + l15) * LDSTRIDE;
            afrag[r] = make_frag(base + sel * 16, base + sel * 16 + 8);
        }
        for (int t = 0; t < 4; ++t) {
            const unsigned short* base = Bt + (wc * 64 + t * 16 + l15) * LDSTRIDE + sel * 16;
            bfrag[t] = make_frag(base, base + 8);
        }
        for (int r = 0; r < 2; ++r)
            for (int t = 0; t < 4; ++t)
                acc[r][t] = __builtin_amdgcn_wmma_f32_16x16x32_bf16(
                    false, afrag[r], false, bfrag[t], (short)0, acc[r][t], false, false);
    }

    // Epilogue (plain stores -> deterministic).
    // Transposed placement block (J,I): per-lane contiguous along crow -> b128s.
    // Off-diagonal: also natural placement block (I,J) as scattered b32.
    for (int r = 0; r < 2; ++r) {
        int crow = ci + wr * 32 + r * 16 + sel * 8;
        for (int t = 0; t < 4; ++t) {
            int dcol = cj + wc * 64 + t * 16 + l15;
            float4 f0, f1;
            f0.x = acc[r][t][0]; f0.y = acc[r][t][1];
            f0.z = acc[r][t][2]; f0.w = acc[r][t][3];
            f1.x = acc[r][t][4]; f1.y = acc[r][t][5];
            f1.z = acc[r][t][6]; f1.w = acc[r][t][7];
            float* pT = part + (size_t)dcol * CCH + crow;   // crow mult of 8 -> aligned
            *(float4*)pT       = f0;
            *(float4*)(pT + 4) = f1;
            if (!diag) {
                for (int v = 0; v < 8; ++v)
                    part[(size_t)(crow + v) * CCH + dcol] = acc[r][t][v];
            }
        }
    }
}

// ---------------------------------------------------------------------------
// Kernel 2: reduce slices + reversed softmax over rows, write transposed bf16.
// attn_T[b][d][c] = softmax_d(min_d x - x)[c][d], x[c][d] = sum_s P[s][b][c][d]
// grid: (CCH/8, BATCH), block 256 (one wave per row).
// ---------------------------------------------------------------------------
__global__ __launch_bounds__(256)
void softmax_kernel(const float* __restrict__ partials, unsigned short* __restrict__ attn_T) {
    const int b    = blockIdx.y;
    const int wid  = threadIdx.x >> 5;
    const int lane = threadIdx.x & 31;
    const int row  = blockIdx.x * 8 + wid;

    float x[16];
    float m = 3.4e38f;
    for (int i = 0; i < 16; ++i) {
        int d = lane + i * 32;
        float s = 0.f;
        for (int sl = 0; sl < NSLICE; ++sl)  // fixed-order, deterministic reduce
            s += partials[((size_t)(sl * BATCH + b) * CCH + row) * CCH + d];
        x[i] = s;
        m = fminf(m, s);
    }
    for (int off = 16; off; off >>= 1) m = fminf(m, __shfl_xor(m, off, 32));

    float p[16], sum = 0.f;
    for (int i = 0; i < 16; ++i) { p[i] = __expf(m - x[i]); sum += p[i]; }
    for (int off = 16; off; off >>= 1) sum += __shfl_xor(sum, off, 32);
    const float inv = 1.0f / sum;

    unsigned short* dst = attn_T + (size_t)b * CCH * CCH;
    for (int i = 0; i < 16; ++i) {
        int d = lane + i * 32;
        dst[(size_t)d * CCH + row] = f2bf(p[i] * inv);
    }
}

// ---------------------------------------------------------------------------
// Kernel 3: out = gamma * (q @ attn) + inputs.  128x128 tile per WG.
// A staged via LDS [n][c]; B fragments direct from global attn_T[d][c] (b128).
// grid: (NPIX/128, CCH/128, BATCH), block 256.
// ---------------------------------------------------------------------------
__global__ __launch_bounds__(256)
void av_kernel(const float* __restrict__ q, const unsigned short* __restrict__ attn_T,
               const float* __restrict__ gamma_p, float* __restrict__ out) {
    __shared__ __align__(16) unsigned short ldsA[128 * LDSTRIDE];  // [n - n0][c]

    const int n0 = blockIdx.x * 128;
    const int d0 = blockIdx.y * 128;
    const int b  = blockIdx.z;

    const float* qb = q + (size_t)b * NPIX * CCH;
    const unsigned short* at = attn_T + (size_t)b * CCH * CCH;
    float* ob = out + (size_t)b * NPIX * CCH;
    const float gamma = gamma_p[0];

    const int tid  = threadIdx.x;
    const int lane = tid & 31;
    const int wid  = tid >> 5;
    const int wr   = wid & 3;
    const int wc   = wid >> 2;
    const int sel  = (lane >> 4) & 1;
    const int l15  = lane & 15;

    v8f acc[2][4];
    for (int r = 0; r < 2; ++r)
        for (int t = 0; t < 4; ++t)
            acc[r][t] = v8f_zero();

    for (int c0 = 0; c0 < CCH; c0 += 32) {
        __syncthreads();
        for (int i = 0; i < 4; ++i) {
            int idx = tid + i * 256;          // 0..1023 over 128n x 8 float4
            int n   = idx >> 3;               // 0..127
            int c4  = (idx & 7) << 2;         // 0..28
            float4 v = *(const float4*)(qb + (size_t)(n0 + n) * CCH + c0 + c4);
            unsigned long long pk = (unsigned long long)pack2bf(v.x, v.y)
                                  | ((unsigned long long)pack2bf(v.z, v.w) << 32);
            *(unsigned long long*)(ldsA + n * LDSTRIDE + c4) = pk;
        }
        __syncthreads();

        v16bf afrag[2], bfrag[4];
        for (int r = 0; r < 2; ++r) {
            const unsigned short* base = ldsA + (wr * 32 + r * 16 + l15) * LDSTRIDE;
            afrag[r] = make_frag(base + sel * 16, base + sel * 16 + 8);
        }
        for (int t = 0; t < 4; ++t) {
            const unsigned short* gp = at + (size_t)(d0 + wc * 64 + t * 16 + l15) * CCH
                                         + c0 + sel * 16;
            bfrag[t] = make_frag(gp, gp + 8);
        }
        for (int r = 0; r < 2; ++r)
            for (int t = 0; t < 4; ++t)
                acc[r][t] = __builtin_amdgcn_wmma_f32_16x16x32_bf16(
                    false, afrag[r], false, bfrag[t], (short)0, acc[r][t], false, false);
    }

    for (int r = 0; r < 2; ++r) {
        int nrow = n0 + wr * 32 + r * 16 + sel * 8;
        for (int t = 0; t < 4; ++t) {
            int d = d0 + wc * 64 + t * 16 + l15;
            for (int v = 0; v < 8; ++v) {
                size_t off = (size_t)(nrow + v) * CCH + d;
                ob[off] = gamma * acc[r][t][v] + qb[off];
            }
        }
    }
}

// ---------------------------------------------------------------------------
extern "C" void kernel_launch(void* const* d_in, const int* in_sizes, int n_in,
                              void* d_out, int out_size, void* d_ws, size_t ws_size,
                              hipStream_t stream) {
    (void)in_sizes; (void)n_in; (void)out_size; (void)ws_size;
    const float* q       = (const float*)d_in[0];
    const float* gamma   = (const float*)d_in[1];
    float*       out     = (float*)d_out;

    const size_t part_elems = (size_t)NSLICE * BATCH * CCH * CCH;  // ~67 MB
    float*          partials = (float*)d_ws;
    unsigned short* attn_T   = (unsigned short*)((char*)d_ws + part_elems * sizeof(float));

    dim3 gg(10, 1, BATCH * NSLICE);                     // upper-triangle block pairs
    gram_kernel<<<gg, 256, 0, stream>>>(q, partials);

    dim3 gs(CCH / 8, BATCH);                            // (64,8)
    softmax_kernel<<<gs, 256, 0, stream>>>(partials, attn_T);

    dim3 gv(NPIX / 128, CCH / 128, BATCH);              // (128,4,8)
    av_kernel<<<gv, 256, 0, stream>>>(q, attn_T, gamma, out);
}
